// CRFMultitaskTagger_25855703122036
// MI455X (gfx1250) — matrix operand order repaired and checked by reading, not compiled, not run on hardware
//
#include <hip/hip_runtime.h>
#include <math.h>

// ---------------------------------------------------------------------------
// BERT-base encoder + CRF heads for MI455X (gfx1250, wave32, WMMA bf16)
//
// Heavy GEMMs run a bf16 pipeline (weights pre-transposed to [n][k] bf16,
// activations produced in bf16 by epilogues). Staging into double-buffered
// LDS uses both CDNA5 data movers:
//   A tile: per-lane global_load_async_to_lds_b128 (ASYNCcnt)
//   B tile: one TDM descriptor per wave0 via tensor_load_to_lds (TENSORcnt),
//           LDS padding fields reproduce the 80B row pitch.
// v_wmma_f32_16x16x32_bf16 consumes the previous tile while both engines
// fill the next one.
// ---------------------------------------------------------------------------

#define BB   16
#define SS   256
#define HH   768
#define LL   12
#define NHH  12
#define HDD  64
#define FFF  3072
#define TMM  9
#define TSS  42
#define TOK  (BB*SS)

typedef __attribute__((ext_vector_type(16))) __bf16 v16bf;
typedef __attribute__((ext_vector_type(8)))  __bf16 v8bf;
typedef __attribute__((ext_vector_type(8)))  float  v8f;
typedef __attribute__((ext_vector_type(4)))  unsigned int u32x4;
typedef __attribute__((ext_vector_type(8)))  int i32x8;
typedef __attribute__((ext_vector_type(4)))  int i32x4;

__device__ __forceinline__ unsigned short bf16bits(float f) {
    union { float f; unsigned u; } a; a.f = f;
    unsigned r = a.u + 0x7fffu + ((a.u >> 16) & 1u);   // RNE
    return (unsigned short)(r >> 16);
}

__device__ __forceinline__ __bf16 f2bf(float f) {
    union { unsigned short s; __bf16 b; } o; o.s = bf16bits(f);
    return o.b;
}

// CDNA5 async copy: 16 bytes global -> LDS, tracked by ASYNCcnt.
__device__ __forceinline__ void async_cp16(unsigned lds, const void* g) {
    asm volatile("global_load_async_to_lds_b128 %0, %1, off"
                 :: "v"(lds), "v"(g) : "memory");
}
__device__ __forceinline__ void wait_async0() {
    asm volatile("s_wait_asynccnt 0x0" ::: "memory");
}

// ---------------------------------------------------------------------------
// TDM: one descriptor moves a 128x32 bf16 tile (rows 64B, global row stride
// strideElems elements) into LDS with 16B padding after each row (80B pitch).
// D# per CDNA5 ISA 8.3/8.4: type=2, count=1, data_size=2B, 2D tile,
// pad_enable, pad_interval=16 DWORDs (code 3), pad_amount=4 DWORDs (code 3).
// This toolchain exposes the 6-arg builtin:
//   (u32x4 g0, i32x8 g1, i32x4 g2, i32x4 g3, i32x8 g4, i32 cpol)
// ---------------------------------------------------------------------------
__device__ __forceinline__ void tdm_load_tile_b(
    unsigned ldsAddr, const void* gaddr, long strideElems)
{
    unsigned long long ga = (unsigned long long)gaddr;
    u32x4 g0;
    g0[0] = 1u;                                        // count=1, user mode
    g0[1] = ldsAddr;                                   // lds_addr (bytes)
    g0[2] = (unsigned)(ga & 0xffffffffu);              // global_addr[31:0]
    g0[3] = (unsigned)((ga >> 32) & 0x01ffffffu)       // global_addr[56:32]
          | (2u << 30);                                // type=2 ("image")
    const unsigned D0 = 0x7fffffffu;                   // tensor_dim0 (huge)
    const unsigned D1 = 0x7fffffffu;                   // tensor_dim1 (huge)
    const unsigned T0 = 32u;                           // tile_dim0 (elems/row)
    const unsigned T1 = 128u;                          // tile_dim1 (rows)
    unsigned long long S0 = (unsigned long long)strideElems;  // dim0 stride
    i32x8 g1;
    g1[0] = (int)((1u << 16)                           // data_size = 2B
                | (1u << 20)                           // pad_enable
                | (3u << 22)                           // pad_interval: 16 DW
                | (3u << 25));                         // pad_amount: 4 DW
    g1[1] = (int)((D0 & 0xffffu) << 16);               // dim0[15:0]
    g1[2] = (int)((D0 >> 16) | ((D1 & 0xffffu) << 16));
    g1[3] = (int)((D1 >> 16) | (T0 << 16));
    g1[4] = (int)T1;                                   // tile_dim1 (tile_dim2=0)
    g1[5] = (int)(S0 & 0xffffffffu);
    g1[6] = (int)((S0 >> 32) & 0xffffu);               // dim1 stride = 0
    g1[7] = 0;
    i32x4 z4; z4[0] = 0; z4[1] = 0; z4[2] = 0; z4[3] = 0;  // groups 2/3 (2D: unused)
    i32x8 z8;
#pragma unroll
    for (int i = 0; i < 8; i++) z8[i] = 0;
    __builtin_amdgcn_tensor_load_to_lds(g0, g1, z4, z4, z8, 0);
}

// ---------------------------------------------------------------------------
// bf16 async/TDM-staged GEMM: C[z] = act(A[z] * B[z] + bias)
//   A bf16 (m,k): A + zoff + m*lda + k          (k contiguous)
//   B bf16 (n,k): B + zoff + n*ldbn + k         (k contiguous, pre-transposed)
//   C f32 or bf16 (m,n): C + zoff + m*ldc + n
// Requires M%128==0, N%128==0, K%32==0, 16B-aligned rows.
// Block 128x128x32, 256 threads = 8 waves (2 M x 4 N), 4x2 WMMA tiles/wave.
// ---------------------------------------------------------------------------
__global__ __launch_bounds__(256) void wmma_gemm_bf16_kernel(
    const unsigned short* __restrict__ A, long lda, long Ao, long Ai,
    const unsigned short* __restrict__ Bm, long ldbn, long Bo, long Bi,
    void* __restrict__ Cv, long ldc, long Co, long Ci,
    const float* __restrict__ bias,
    int M, int N, int K, int innerCount, int actGelu, int outBf)
{
    __shared__ __bf16 As[2][128][40];   // [buf][m][k], 80B row pitch (16B aligned)
    __shared__ __bf16 Bs[2][128][40];   // [buf][n][k]

    const int tid  = threadIdx.x;
    const int lane = tid & 31;
    const int wave = tid >> 5;
    const int wm   = wave & 1;
    const int wn   = wave >> 1;

    const int z  = blockIdx.z;
    const int bo = z / innerCount;
    const int bi = z - bo * innerCount;
    A  += (long)bo * Ao + (long)bi * Ai;
    Bm += (long)bo * Bo + (long)bi * Bi;

    const int m0 = blockIdx.y * 128;
    const int n0 = blockIdx.x * 128;

    v8f acc[4][2];
#pragma unroll
    for (int i = 0; i < 4; i++)
#pragma unroll
        for (int j = 0; j < 2; j++)
#pragma unroll
            for (int e = 0; e < 8; e++) acc[i][j][e] = 0.0f;

    const int half = lane >> 4;
    const int r16  = lane & 15;
    const int ak0  = half ? 8 : 0;     // A frag: K 0-7/16-23 vs 8-15/24-31
    const int bk0  = half ? 8 : 0;     // B frag: K 0-7/16-23 vs 8-15/24-31

    // Fill one LDS buffer: A by per-lane async copies, B by one TDM op.
    auto issue = [&](int buf, int k0) {
#pragma unroll
        for (int i = 0; i < 2; i++) {
            int chunk = tid + i * 256;          // 0..511
            int row = chunk >> 2;               // 0..127
            int c16 = (chunk & 3) * 8;          // bf16 element offset (16B)
            async_cp16((unsigned)(unsigned long long)&As[buf][row][c16],
                       A + (long)(m0 + row) * lda + k0 + c16);
        }
        if (wave == 0) {
            tdm_load_tile_b((unsigned)(unsigned long long)&Bs[buf][0][0],
                            Bm + (long)n0 * ldbn + k0, ldbn);
        }
        if (k0 + 64 < K) {  // pull the tile after next into L2
            __builtin_prefetch(A + (long)(m0 + (tid >> 1)) * lda + k0 + 64 + (tid & 1) * 16, 0, 1);
            __builtin_prefetch(Bm + (long)(n0 + (tid >> 1)) * ldbn + k0 + 64 + (tid & 1) * 16, 0, 1);
        }
    };
    auto wait_tiles = [&]() {
        wait_async0();
        __builtin_amdgcn_s_wait_tensorcnt(0);
    };

    auto compute = [&](int buf) {
        v16bf afrag[4];
#pragma unroll
        for (int mt = 0; mt < 4; mt++) {
            int mrow = wm * 64 + mt * 16 + r16;
            v8bf lo = *(const v8bf*)&As[buf][mrow][ak0];
            v8bf hi = *(const v8bf*)&As[buf][mrow][ak0 + 16];
            v16bf a;
#pragma unroll
            for (int e = 0; e < 8; e++) { a[e] = lo[e]; a[e + 8] = hi[e]; }
            afrag[mt] = a;
        }
#pragma unroll
        for (int nt = 0; nt < 2; nt++) {
            int nrow = wn * 32 + nt * 16 + r16;
            v8bf lo = *(const v8bf*)&Bs[buf][nrow][bk0];
            v8bf hi = *(const v8bf*)&Bs[buf][nrow][bk0 + 16];
            v16bf bfrag;
#pragma unroll
            for (int e = 0; e < 8; e++) { bfrag[e] = lo[e]; bfrag[e + 8] = hi[e]; }
#pragma unroll
            for (int mt = 0; mt < 4; mt++) {
                acc[mt][nt] = __builtin_amdgcn_wmma_f32_16x16x32_bf16(
                    false, afrag[mt], false, bfrag, (short)0, acc[mt][nt], false, false);
            }
        }
    };

    issue(0, 0);
    wait_tiles();
    __syncthreads();
    int buf = 0;
    for (int k0 = 0; k0 < K; k0 += 32) {
        bool more = (k0 + 32) < K;
        if (more) issue(buf ^ 1, k0 + 32);   // both engines fill next buffer
        compute(buf);                         // WMMA on current buffer
        if (more) wait_tiles();
        __syncthreads();
        buf ^= 1;
    }

    // Epilogue: 16x16 f32 D map (lane%16 = N, VGPR r -> M = r + 8*(lane>=16)).
#pragma unroll
    for (int mt = 0; mt < 4; mt++) {
#pragma unroll
        for (int nt = 0; nt < 2; nt++) {
            int gn = n0 + wn * 32 + nt * 16 + r16;
            float bv = bias ? bias[gn] : 0.0f;
#pragma unroll
            for (int r = 0; r < 8; r++) {
                int gm = m0 + wm * 64 + mt * 16 + (half ? 8 : 0) + r;
                float v = acc[mt][nt][r] + bv;
                if (actGelu) v = 0.5f * v * (1.0f + erff(v * 0.70710678118f));
                long off = ((long)bo * Co + (long)bi * Ci) + (long)gm * ldc + gn;
                if (outBf) ((unsigned short*)Cv)[off] = bf16bits(v);
                else       ((float*)Cv)[off] = v;
            }
        }
    }
}

// ---------------------------------------------------------------------------
// Guarded f32 GEMM for ragged shapes (ctx N=64, heads N=9/42).
// ---------------------------------------------------------------------------
__global__ __launch_bounds__(256) void wmma_gemm_f32_kernel(
    const float* __restrict__ A, long lda, long Ao, long Ai,
    const float* __restrict__ Bm, long ldbk, long ldbn, long Bo, long Bi,
    void* __restrict__ Cv, long ldc, long Co, long Ci,
    const float* __restrict__ bias,
    int M, int N, int K, int innerCount, int actGelu, int outBf)
{
    __shared__ __bf16 As[128][40];
    __shared__ __bf16 Bs[128][40];

    const int tid  = threadIdx.x;
    const int lane = tid & 31;
    const int wave = tid >> 5;
    const int wm   = wave & 1;
    const int wn   = wave >> 1;

    const int z  = blockIdx.z;
    const int bo = z / innerCount;
    const int bi = z - bo * innerCount;
    A  += (long)bo * Ao + (long)bi * Ai;
    Bm += (long)bo * Bo + (long)bi * Bi;

    const int m0 = blockIdx.y * 128;
    const int n0 = blockIdx.x * 128;

    v8f acc[4][2];
#pragma unroll
    for (int i = 0; i < 4; i++)
#pragma unroll
        for (int j = 0; j < 2; j++)
#pragma unroll
            for (int e = 0; e < 8; e++) acc[i][j][e] = 0.0f;

    const int half = lane >> 4;
    const int r16  = lane & 15;
    const int ak0  = half ? 8 : 0;
    const int bk0  = half ? 8 : 0;

    for (int k0 = 0; k0 < K; k0 += 32) {
#pragma unroll
        for (int i = 0; i < 16; i++) {
            int idx = tid + i * 256;
            int rr = idx >> 5, cc = idx & 31;
            int gk = k0 + cc;
            int gm = m0 + rr;
            float va = (gm < M && gk < K) ? A[(long)gm * lda + gk] : 0.0f;
            As[rr][cc] = f2bf(va);
            int gn = n0 + rr;
            float vb = (gn < N && gk < K) ? Bm[(long)gk * ldbk + (long)gn * ldbn] : 0.0f;
            Bs[rr][cc] = f2bf(vb);
        }
        __syncthreads();
        v16bf afrag[4];
#pragma unroll
        for (int mt = 0; mt < 4; mt++) {
            int mrow = wm * 64 + mt * 16 + r16;
            v8bf lo = *(const v8bf*)&As[mrow][ak0];
            v8bf hi = *(const v8bf*)&As[mrow][ak0 + 16];
            v16bf a;
#pragma unroll
            for (int e = 0; e < 8; e++) { a[e] = lo[e]; a[e + 8] = hi[e]; }
            afrag[mt] = a;
        }
#pragma unroll
        for (int nt = 0; nt < 2; nt++) {
            int nrow = wn * 32 + nt * 16 + r16;
            v8bf lo = *(const v8bf*)&Bs[nrow][bk0];
            v8bf hi = *(const v8bf*)&Bs[nrow][bk0 + 16];
            v16bf bfrag;
#pragma unroll
            for (int e = 0; e < 8; e++) { bfrag[e] = lo[e]; bfrag[e + 8] = hi[e]; }
#pragma unroll
            for (int mt = 0; mt < 4; mt++) {
                acc[mt][nt] = __builtin_amdgcn_wmma_f32_16x16x32_bf16(
                    false, afrag[mt], false, bfrag, (short)0, acc[mt][nt], false, false);
            }
        }
        __syncthreads();
    }

#pragma unroll
    for (int mt = 0; mt < 4; mt++) {
#pragma unroll
        for (int nt = 0; nt < 2; nt++) {
            int gn = n0 + wn * 32 + nt * 16 + r16;
            if (gn >= N) continue;
            float bv = bias ? bias[gn] : 0.0f;
#pragma unroll
            for (int r = 0; r < 8; r++) {
                int gm = m0 + wm * 64 + mt * 16 + (half ? 8 : 0) + r;
                if (gm >= M) continue;
                float v = acc[mt][nt][r] + bv;
                if (actGelu) v = 0.5f * v * (1.0f + erff(v * 0.70710678118f));
                long off = ((long)bo * Co + (long)bi * Ci) + (long)gm * ldc + gn;
                if (outBf) ((unsigned short*)Cv)[off] = bf16bits(v);
                else       ((float*)Cv)[off] = v;
            }
        }
    }
}

// ---------------------------------------------------------------------------
// Weight transpose + f32->bf16: src (K x N, row-major) -> dst (N x K) bf16.
// ---------------------------------------------------------------------------
__global__ __launch_bounds__(256) void transpose_bf16_kernel(
    const float* __restrict__ src, unsigned short* __restrict__ dst,
    int K, int N)
{
    __shared__ float tile[32][33];
    const int kb = blockIdx.y * 32, nb = blockIdx.x * 32;
    const int tx = threadIdx.x & 31, ty = threadIdx.x >> 5;   // 32 x 8
#pragma unroll
    for (int i = 0; i < 4; i++)
        tile[ty + i * 8][tx] = src[(long)(kb + ty + i * 8) * N + nb + tx];
    __syncthreads();
#pragma unroll
    for (int i = 0; i < 4; i++)
        dst[(long)(nb + ty + i * 8) * K + kb + tx] = bf16bits(tile[tx][ty + i * 8]);
}

// ---------------------------------------------------------------------------
__global__ __launch_bounds__(256) void embed_kernel(
    const int* __restrict__ ids, const float* __restrict__ we,
    const float* __restrict__ pe, const float* __restrict__ te,
    float* __restrict__ out)
{
    long tok = blockIdx.x;
    int s = (int)(tok % SS);
    int id = ids[tok];
    int t = threadIdx.x;
#pragma unroll
    for (int i = 0; i < 3; i++) {
        int c = t + i * 256;
        out[tok * HH + c] = we[(long)id * HH + c] + pe[(long)s * HH + c] + te[c];
    }
}

// ---------------------------------------------------------------------------
__global__ __launch_bounds__(256) void add_ln_kernel(
    const float* __restrict__ x, const float* __restrict__ res,
    const float* __restrict__ gam, const float* __restrict__ bet,
    float* __restrict__ out, unsigned short* __restrict__ outB)
{
    const long row = blockIdx.x;
    const int t = threadIdx.x;
    __shared__ float rsum[256], rsq[256];
    float v[3];
    float s = 0.f, s2 = 0.f;
#pragma unroll
    for (int i = 0; i < 3; i++) {
        int c = t + i * 256;
        float a = x[row * HH + c];
        if (res) a += res[row * HH + c];
        v[i] = a; s += a; s2 += a * a;
    }
    rsum[t] = s; rsq[t] = s2;
    __syncthreads();
    for (int st = 128; st > 0; st >>= 1) {
        if (t < st) { rsum[t] += rsum[t + st]; rsq[t] += rsq[t + st]; }
        __syncthreads();
    }
    float mean = rsum[0] * (1.0f / HH);
    float var  = rsq[0] * (1.0f / HH) - mean * mean;
    float inv  = rsqrtf(var + 1e-12f);
#pragma unroll
    for (int i = 0; i < 3; i++) {
        int c = t + i * 256;
        float y = (v[i] - mean) * inv * gam[c] + bet[c];
        out[row * HH + c] = y;
        if (outB) outB[row * HH + c] = bf16bits(y);
    }
}

// ---------------------------------------------------------------------------
__global__ __launch_bounds__(256) void softmax_kernel(
    float* __restrict__ sc, const int* __restrict__ am)
{
    const int q = blockIdx.x;
    const int z = blockIdx.y;
    const int b = z / NHH;
    float* row = sc + (long)z * SS * SS + (long)q * SS;
    const int t = threadIdx.x;
    __shared__ float red[256];
    float x = row[t] * 0.125f + (1.0f - (float)am[b * SS + t]) * (-1.0e9f);
    red[t] = x;
    __syncthreads();
    for (int s = 128; s > 0; s >>= 1) {
        if (t < s) red[t] = fmaxf(red[t], red[t + s]);
        __syncthreads();
    }
    float mx = red[0];
    __syncthreads();
    float e = expf(x - mx);
    red[t] = e;
    __syncthreads();
    for (int s = 128; s > 0; s >>= 1) {
        if (t < s) red[t] += red[t + s];
        __syncthreads();
    }
    row[t] = e / red[0];
}

// ---------------------------------------------------------------------------
__global__ void crf_lognorm_kernel(
    const float* __restrict__ em, const int* __restrict__ fsw,
    const float* __restrict__ st, const float* __restrict__ en,
    const float* __restrict__ tr, float* __restrict__ out)
{
    const int b = blockIdx.x;
    const int t = threadIdx.x;
    __shared__ float sc[TMM], ns[TMM];
    if (t < TMM) sc[t] = st[t] + em[((long)b * SS) * TMM + t];
    __syncthreads();
    for (int step = 1; step < SS; step++) {
        if (t < TMM) {
            float mx = -3.4e38f;
            for (int i = 0; i < TMM; i++) mx = fmaxf(mx, sc[i] + tr[i * TMM + t]);
            float sum = 0.f;
            for (int i = 0; i < TMM; i++) sum += expf(sc[i] + tr[i * TMM + t] - mx);
            ns[t] = mx + logf(sum) + em[((long)b * SS + step) * TMM + t];
        }
        __syncthreads();
        bool m = fsw[b * SS + step] != 0;
        if (m && t < TMM) sc[t] = ns[t];
        __syncthreads();
    }
    if (t == 0) {
        float mx = -3.4e38f;
        for (int j = 0; j < TMM; j++) mx = fmaxf(mx, sc[j] + en[j]);
        float sum = 0.f;
        for (int j = 0; j < TMM; j++) sum += expf(sc[j] + en[j] - mx);
        out[b] = mx + logf(sum);
    }
}

// ---------------------------------------------------------------------------
__global__ void crf_score_kernel(
    const float* __restrict__ em, const int* __restrict__ tags,
    const int* __restrict__ fsw, const float* __restrict__ st,
    const float* __restrict__ en, const float* __restrict__ tr,
    float* __restrict__ out)
{
    int b = threadIdx.x;
    if (b >= BB) return;
    const int* tg = tags + b * SS;
    float sc = st[tg[0]] + em[((long)b * SS) * TMM + tg[0]];
    int cnt = 1;
    for (int t = 1; t < SS; t++) {
        bool m = fsw[b * SS + t] != 0;
        if (m) sc += tr[tg[t - 1] * TMM + tg[t]] + em[((long)b * SS + t) * TMM + tg[t]];
        cnt += m ? 1 : 0;
    }
    out[b] = sc + en[tg[cnt - 1]];
}

// ---------------------------------------------------------------------------
__global__ void viterbi_kernel(
    const float* __restrict__ em, const int* __restrict__ fsw,
    const float* __restrict__ st, const float* __restrict__ en,
    const float* __restrict__ tr, int* __restrict__ hist,
    float* __restrict__ outTags)
{
    const int b = blockIdx.x;
    const int t = threadIdx.x;
    __shared__ float sc[TMM], ns[TMM];
    if (t < TMM) sc[t] = st[t] + em[((long)b * SS) * TMM + t];
    __syncthreads();
    for (int step = 1; step < SS; step++) {
        if (t < TMM) {
            float best = -3.4e38f; int idx = 0;
            for (int i = 0; i < TMM; i++) {
                float c = sc[i] + tr[i * TMM + t];
                if (c > best) { best = c; idx = i; }
            }
            ns[t] = best + em[((long)b * SS + step) * TMM + t];
            hist[((long)b * (SS - 1) + (step - 1)) * TMM + t] = idx;
        }
        __syncthreads();
        bool m = fsw[b * SS + step] != 0;
        if (m && t < TMM) sc[t] = ns[t];
        __syncthreads();
    }
    if (t == 0) {
        int cnt = 1;
        for (int s2 = 1; s2 < SS; s2++) cnt += (fsw[b * SS + s2] != 0) ? 1 : 0;
        int seqEnd = cnt - 1;
        float best = -3.4e38f; int cur = 0;
        for (int j = 0; j < TMM; j++) {
            float c = sc[j] + en[j];
            if (c > best) { best = c; cur = j; }
        }
        outTags[(long)b * SS + SS - 1] = (float)cur;
        for (int s2 = SS - 2; s2 >= 0; s2--) {
            int prev = hist[((long)b * (SS - 1) + s2) * TMM + cur];
            if (s2 < seqEnd) cur = prev;
            outTags[(long)b * SS + s2] = (float)cur;
        }
    }
}

// ---------------------------------------------------------------------------
__global__ __launch_bounds__(256) void sup_kernel(
    const float* __restrict__ logits, const int* __restrict__ tags,
    float* __restrict__ preds, float* __restrict__ nllArr,
    float* __restrict__ validArr)
{
    int tok = blockIdx.x * 256 + threadIdx.x;
    if (tok >= TOK) return;
    const float* rw = logits + (long)tok * TSS;
    float mx = rw[0]; int am = 0;
    for (int j = 1; j < TSS; j++) if (rw[j] > mx) { mx = rw[j]; am = j; }
    float sum = 0.f;
    for (int j = 0; j < TSS; j++) sum += expf(rw[j] - mx);
    float lse = mx + logf(sum);
    preds[tok] = (float)am;
    int tg = tags[tok];
    bool valid = (tg != -100);
    int t2 = valid ? tg : 0;
    nllArr[tok]   = valid ? (lse - rw[t2]) : 0.0f;
    validArr[tok] = valid ? 1.0f : 0.0f;
}

// ---------------------------------------------------------------------------
__global__ void combine_kernel(
    const float* __restrict__ crfS, const float* __restrict__ crfN,
    const float* __restrict__ nllArr, const float* __restrict__ validArr,
    float* __restrict__ out0)
{
    if (threadIdx.x != 0 || blockIdx.x != 0) return;
    float s = 0.f;
    for (int b = 0; b < BB; b++) s += crfS[b] - crfN[b];
    float mwe = -s / (float)BB;
    double nll = 0.0, cnt = 0.0;
    for (int i = 0; i < TOK; i++) { nll += nllArr[i]; cnt += validArr[i]; }
    float c = (float)cnt; if (c < 1.0f) c = 1.0f;
    out0[0] = mwe + (float)(nll) / c;
}

// ---------------------------------------------------------------------------
extern "C" void kernel_launch(void* const* d_in, const int* in_sizes, int n_in,
                              void* d_out, int out_size, void* d_ws, size_t ws_size,
                              hipStream_t stream)
{
    const int*   ids      = (const int*)  d_in[0];
    const int*   am       = (const int*)  d_in[1];
    const int*   fsw      = (const int*)  d_in[2];
    const int*   mwe_tags = (const int*)  d_in[3];
    const int*   sup_tags = (const int*)  d_in[4];
    const float* word_emb = (const float*)d_in[5];
    const float* pos_emb  = (const float*)d_in[6];
    const float* type_emb = (const float*)d_in[7];
    const float* eln_s    = (const float*)d_in[8];
    const float* eln_b    = (const float*)d_in[9];
    const float* Wq = (const float*)d_in[10]; const float* bq = (const float*)d_in[11];
    const float* Wk = (const float*)d_in[12]; const float* bk = (const float*)d_in[13];
    const float* Wv = (const float*)d_in[14]; const float* bv = (const float*)d_in[15];
    const float* Wo = (const float*)d_in[16]; const float* bo = (const float*)d_in[17];
    const float* l1s = (const float*)d_in[18]; const float* l1b = (const float*)d_in[19];
    const float* Wf1 = (const float*)d_in[20]; const float* bf1 = (const float*)d_in[21];
    const float* Wf2 = (const float*)d_in[22]; const float* bf2 = (const float*)d_in[23];
    const float* l2s = (const float*)d_in[24]; const float* l2b = (const float*)d_in[25];
    const float* mweW = (const float*)d_in[26]; const float* mweB = (const float*)d_in[27];
    const float* supW = (const float*)d_in[28]; const float* supB = (const float*)d_in[29];
    const float* crf_st = (const float*)d_in[30];
    const float* crf_en = (const float*)d_in[31];
    const float* crf_tr = (const float*)d_in[32];

    float* out = (float*)d_out;   // [loss, mwe_preds(4096), sup_preds(4096)]

    // ---- workspace carving (256B aligned chunks) ----
    char* base = (char*)d_ws;
    auto alloc = [&](size_t bytes) -> void* {
        void* p = (void*)base; base += (bytes + 255) & ~(size_t)255; return p;
    };
    float* h    = (float*)alloc((size_t)TOK * HH * 4);
    float* tmp  = (float*)alloc((size_t)TOK * HH * 4);
    float* vbuf = (float*)alloc((size_t)TOK * HH * 4);
    float* big  = (float*)alloc((size_t)TOK * FFF * 4);   // attn scores/probs
    float* mweL = (float*)alloc((size_t)TOK * TMM * 4);
    float* supL = (float*)alloc((size_t)TOK * TSS * 4);
    float* crfN = (float*)alloc(64);
    float* crfS = (float*)alloc(64);
    float* nllA = (float*)alloc((size_t)TOK * 4);
    float* valA = (float*)alloc((size_t)TOK * 4);
    int*   hist = (int*)  alloc((size_t)BB * (SS - 1) * TMM * 4);
    unsigned short* hB   = (unsigned short*)alloc((size_t)TOK * HH * 2);
    unsigned short* qB   = (unsigned short*)alloc((size_t)TOK * HH * 2);
    unsigned short* kB   = (unsigned short*)alloc((size_t)TOK * HH * 2);
    unsigned short* ctxB = (unsigned short*)alloc((size_t)TOK * HH * 2);
    unsigned short* bigB = (unsigned short*)alloc((size_t)TOK * FFF * 2);
    unsigned short* wB   = (unsigned short*)alloc((size_t)7077888 * 2);  // per-layer [n][k]
    const long WQO = 0, WKO = 589824, WVO = 2 * 589824, WOO = 3 * 589824;
    const long WF1O = 4L * 589824, WF2O = 4L * 589824 + (long)HH * FFF;

    auto GEMMB = [&](const unsigned short* A, long lda, long Ao, long Ai,
                     const unsigned short* Bp, long lbn, long Bo, long Bi,
                     void* Cp, long ldc, long Co, long Ci, const float* bias,
                     int M, int N, int K, int inner, int batches, int gelu, int outBf) {
        dim3 g(N / 128, M / 128, batches);
        wmma_gemm_bf16_kernel<<<g, 256, 0, stream>>>(
            A, lda, Ao, Ai, Bp, lbn, Bo, Bi, Cp, ldc, Co, Ci, bias,
            M, N, K, inner, gelu, outBf);
    };
    auto GEMMF = [&](const float* A, long lda, long Ao, long Ai,
                     const float* Bp, long lbk, long lbn, long Bo, long Bi,
                     void* Cp, long ldc, long Co, long Ci, const float* bias,
                     int M, int N, int K, int inner, int batches, int gelu, int outBf) {
        dim3 g((N + 127) / 128, (M + 127) / 128, batches);
        wmma_gemm_f32_kernel<<<g, 256, 0, stream>>>(
            A, lda, Ao, Ai, Bp, lbk, lbn, Bo, Bi, Cp, ldc, Co, Ci, bias,
            M, N, K, inner, gelu, outBf);
    };
    auto TRANS = [&](const float* src, unsigned short* dst, int K, int N) {
        transpose_bf16_kernel<<<dim3(N / 32, K / 32), 256, 0, stream>>>(src, dst, K, N);
    };

    // Embedding + LN (writes h f32 + hB bf16)
    embed_kernel<<<TOK, 256, 0, stream>>>(ids, word_emb, pos_emb, type_emb, tmp);
    add_ln_kernel<<<TOK, 256, 0, stream>>>(tmp, nullptr, eln_s, eln_b, h, hB);

    for (int l = 0; l < LL; l++) {
        long wOff = (long)l * HH * HH, bOff = (long)l * HH;
        TRANS(Wq + wOff, wB + WQO, HH, HH);
        TRANS(Wk + wOff, wB + WKO, HH, HH);
        TRANS(Wv + wOff, wB + WVO, HH, HH);
        TRANS(Wo + wOff, wB + WOO, HH, HH);
        TRANS(Wf1 + (long)l * HH * FFF, wB + WF1O, HH, FFF);
        TRANS(Wf2 + (long)l * FFF * HH, wB + WF2O, FFF, HH);

        // QKV projections: q,k -> bf16; v -> f32 (for ctx GEMM)
        GEMMB(hB, HH, 0, 0, wB + WQO, HH, 0, 0, qB,   HH, 0, 0, bq + bOff, TOK, HH, HH, 1, 1, 0, 1);
        GEMMB(hB, HH, 0, 0, wB + WKO, HH, 0, 0, kB,   HH, 0, 0, bk + bOff, TOK, HH, HH, 1, 1, 0, 1);
        GEMMB(hB, HH, 0, 0, wB + WVO, HH, 0, 0, vbuf, HH, 0, 0, bv + bOff, TOK, HH, HH, 1, 1, 0, 0);
        // scores[b,h] = Q @ K^T (batched 192x: 256x256x64)
        GEMMB(qB, HH, (long)SS * HH, HDD,
              kB, HH, (long)SS * HH, HDD,
              big, SS, (long)NHH * SS * SS, (long)SS * SS,
              nullptr, SS, SS, HDD, NHH, BB * NHH, 0, 0);
        softmax_kernel<<<dim3(SS, BB * NHH), 256, 0, stream>>>(big, am);
        // ctx[b,h] = P @ V (guarded f32: N=64) -> bf16 ctxB
        GEMMF(big, SS, (long)NHH * SS * SS, (long)SS * SS,
              vbuf, HH, 1, (long)SS * HH, HDD,
              ctxB, HH, (long)SS * HH, HDD,
              nullptr, SS, HDD, SS, NHH, BB * NHH, 0, 1);
        // Output projection + residual LN
        GEMMB(ctxB, HH, 0, 0, wB + WOO, HH, 0, 0, tmp, HH, 0, 0, bo + bOff, TOK, HH, HH, 1, 1, 0, 0);
        add_ln_kernel<<<TOK, 256, 0, stream>>>(tmp, h, l1s + bOff, l1b + bOff, h, hB);
        // FFN
        GEMMB(hB, HH, 0, 0, wB + WF1O, HH, 0, 0, bigB, FFF, 0, 0, bf1 + (long)l * FFF,
              TOK, FFF, HH, 1, 1, 1, 1);
        GEMMB(bigB, FFF, 0, 0, wB + WF2O, FFF, 0, 0, tmp, HH, 0, 0, bf2 + bOff,
              TOK, HH, FFF, 1, 1, 0, 0);
        add_ln_kernel<<<TOK, 256, 0, stream>>>(tmp, h, l2s + bOff, l2b + bOff, h, hB);
    }

    // Classifier heads (guarded f32: N=9 / N=42)
    GEMMF(h, HH, 0, 0, mweW, TMM, 1, 0, 0, mweL, TMM, 0, 0, mweB, TOK, TMM, HH, 1, 1, 0, 0);
    GEMMF(h, HH, 0, 0, supW, TSS, 1, 0, 0, supL, TSS, 0, 0, supB, TOK, TSS, HH, 1, 1, 0, 0);

    // CRF + sup losses / predictions
    crf_lognorm_kernel<<<BB, 32, 0, stream>>>(mweL, fsw, crf_st, crf_en, crf_tr, crfN);
    crf_score_kernel<<<1, 32, 0, stream>>>(mweL, mwe_tags, fsw, crf_st, crf_en, crf_tr, crfS);
    viterbi_kernel<<<BB, 32, 0, stream>>>(mweL, fsw, crf_st, crf_en, crf_tr, hist, out + 1);
    sup_kernel<<<(TOK + 255) / 256, 256, 0, stream>>>(supL, sup_tags, out + 1 + TOK, nllA, valA);
    combine_kernel<<<1, 32, 0, stream>>>(crfS, crfN, nllA, valA, out);
}
